// QLLKTransformer_84834194031188
// MI455X (gfx1250) — matrix operand order repaired and checked by reading, not compiled, not run on hardware
//
#include <hip/hip_runtime.h>
#include <hip/hip_bf16.h>

typedef __attribute__((ext_vector_type(16))) _Float16 v16h;
typedef __attribute__((ext_vector_type(8)))  float    v8f;
typedef _Float16 h16;

#define DIM 512
#define NL  4
#define PP  8
#define BB  4
#define LL  8192
#define LP  1024
#define ROWS (BB*LP)   // 4096 patch rows

// ---------------------------------------------------------------------------
// Tiled WMMA GEMM:  C[M,N] = A[M,K](f16) * B[K,N](f16, given as Bt[N,K])
// Block = 256 threads (8 waves), tile 128x128, k-step 32.
// Each wave computes a 32x64 sub-tile: 2 A-frags x 4 B-frags = 8 WMMA/k-step,
// all fragments preloaded so the 8 v_wmma issue back-to-back.
// Epilogue: + bias[col % bias_mod] (if bias) and += old C (if acc_flag).
// ---------------------------------------------------------------------------
__global__ __launch_bounds__(256)
void k_wmma_gemm(const h16* __restrict__ A, int lda,
                 const h16* __restrict__ Bt, int ldb,
                 float* __restrict__ C, int ldc, int K,
                 const float* __restrict__ bias, int bias_mod, int acc_flag)
{
    __shared__ h16 As[128 * 32];   // 8 KB
    __shared__ h16 Bs[128 * 32];   // 8 KB

    const int tid  = threadIdx.x;
    const int wave = tid >> 5;
    const int lane = tid & 31;
    const int mblk = wave >> 1;    // 0..3 : rows  [mblk*32, +32)
    const int nblk = wave & 1;     // 0..1 : cols  [nblk*64, +64)
    const int m0   = blockIdx.y * 128;
    const int n0   = blockIdx.x * 128;

    v8f acc[2][4];
#pragma unroll
    for (int mt = 0; mt < 2; ++mt)
#pragma unroll
        for (int t = 0; t < 4; ++t)
#pragma unroll
            for (int r = 0; r < 8; ++r) acc[mt][t][r] = 0.0f;

    const h16* Ab = A  + (size_t)m0 * lda;
    const h16* Bb = Bt + (size_t)n0 * ldb;

    const int lrow = lane & 15;
    const int half = lane >> 4;    // 0 or 1

    for (int k0 = 0; k0 < K; k0 += 32) {
        // Stage tiles: 128 rows x 32 halfs each = 8KB; 2x 16B chunks per thread.
#pragma unroll
        for (int c = 0; c < 2; ++c) {
            int ch  = tid + c * 256;     // 0..511
            int row = ch >> 2;           // 0..127
            int cc  = ch & 3;            // 16B chunk in 64B row
            *(uint4*)&As[row * 32 + cc * 8] =
                *(const uint4*)&Ab[(size_t)row * lda + k0 + cc * 8];
            *(uint4*)&Bs[row * 32 + cc * 8] =
                *(const uint4*)&Bb[(size_t)row * ldb + k0 + cc * 8];
        }
        // Prefetch next K tile (global_prefetch_b8, near scope).
        if (k0 + 32 < K) {
            int row = tid >> 1, cc = tid & 1;
            __builtin_prefetch(&Ab[(size_t)row * lda + k0 + 32 + cc * 16], 0, 3);
            __builtin_prefetch(&Bb[(size_t)row * ldb + k0 + 32 + cc * 16], 0, 3);
        }
        __syncthreads();

        // Preload all fragments, then burst 8 WMMAs.
        union { v16h v; unsigned u[8]; } a[2], b[4];
#pragma unroll
        for (int mt = 0; mt < 2; ++mt) {
            int arow = mblk * 32 + mt * 16 + lrow;
            // A 16x32 f16 layout: lanes 0-15 -> K {0..7,16..23}; 16-31 -> {8..15,24..31}
#pragma unroll
            for (int p = 0; p < 8; ++p) {
                int kk = half * 8 + ((p < 4) ? (p * 2) : (16 + (p - 4) * 2));
                a[mt].u[p] = *(const unsigned*)&As[arow * 32 + kk];
            }
        }
#pragma unroll
        for (int t = 0; t < 4; ++t) {
            int brow = nblk * 64 + t * 16 + lrow;
            // B 32x16 layout: lanes 0-15 -> K {0..7,16..23}; 16-31 -> {8..15,24..31}
#pragma unroll
            for (int p = 0; p < 8; ++p) {
                int kk = half * 8 + ((p < 4) ? (p * 2) : (16 + (p - 4) * 2));
                b[t].u[p] = *(const unsigned*)&Bs[brow * 32 + kk];
            }
        }
#pragma unroll
        for (int mt = 0; mt < 2; ++mt)
#pragma unroll
            for (int t = 0; t < 4; ++t)
                acc[mt][t] = __builtin_amdgcn_wmma_f32_16x16x32_f16(
                    false, a[mt].v, false, b[t].v, (short)0, acc[mt][t],
                    false, false);
        __syncthreads();
    }

    // Epilogue. C/D layout: lane -> N = lane&15, VGPR r -> M = r + 8*(lane>>4).
#pragma unroll
    for (int mt = 0; mt < 2; ++mt) {
#pragma unroll
        for (int t = 0; t < 4; ++t) {
            int col = n0 + nblk * 64 + t * 16 + lrow;
            float bv = bias ? bias[col % bias_mod] : 0.0f;
#pragma unroll
            for (int r = 0; r < 8; ++r) {
                int row = m0 + mblk * 32 + mt * 16 + half * 8 + r;
                size_t off = (size_t)row * ldc + col;
                float v = acc[mt][t][r] + bv;
                if (acc_flag) v += C[off];
                C[off] = v;
            }
        }
    }
}

// ---------------------------------------------------------------------------
// Elementwise / helper kernels
// ---------------------------------------------------------------------------
__global__ void k_cvt(const float* __restrict__ s, h16* __restrict__ d, int n) {
    int i = blockIdx.x * blockDim.x + threadIdx.x;
    if (i < n) d[i] = (h16)s[i];
}

// W[K,N] (f32, row-major) -> Wt[N,K] (f16)
__global__ void k_cvt_t(const float* __restrict__ s, h16* __restrict__ d,
                        int K, int N) {
    int i = blockIdx.x * blockDim.x + threadIdx.x;
    if (i < K * N) {
        int n = i / K, k = i - n * K;
        d[i] = (h16)s[(size_t)k * N + n];
    }
}

// Gathered byte embedding as f16 GEMM-A: A[r, p*512+d] = emb[bytes[r*8+p], d]
__global__ void k_embed(const int* __restrict__ bytes,
                        const float* __restrict__ emb, h16* __restrict__ A) {
    int i = blockIdx.x * blockDim.x + threadIdx.x;   // 0 .. 4096*4096-1
    int r = i >> 12, c = i & 4095;
    int p = c >> 9,  dd = c & 511;
    A[i] = (h16)emb[(size_t)bytes[r * PP + p] * DIM + dd];
}

__global__ __launch_bounds__(256)
void k_layernorm(const float* __restrict__ x, const float* __restrict__ g,
                 const float* __restrict__ b, float* __restrict__ h) {
    __shared__ float red[256];
    int row = blockIdx.x, t = threadIdx.x;
    const float* xr = x + (size_t)row * DIM;
    float v0 = xr[t], v1 = xr[t + 256];
    red[t] = v0 + v1; __syncthreads();
    for (int s = 128; s > 0; s >>= 1) {
        if (t < s) red[t] += red[t + s];
        __syncthreads();
    }
    float mean = red[0] * (1.0f / DIM); __syncthreads();
    float d0 = v0 - mean, d1 = v1 - mean;
    red[t] = d0 * d0 + d1 * d1; __syncthreads();
    for (int s = 128; s > 0; s >>= 1) {
        if (t < s) red[t] += red[t + s];
        __syncthreads();
    }
    float rstd = rsqrtf(red[0] * (1.0f / DIM) + 1e-5f);
    h[(size_t)row * DIM + t]       = d0 * rstd * g[t]       + b[t];
    h[(size_t)row * DIM + t + 256] = d1 * rstd * g[t + 256] + b[t + 256];
}

__global__ void k_tanh_add(float* __restrict__ x, const float* __restrict__ t, int n) {
    int i = blockIdx.x * blockDim.x + threadIdx.x;
    if (i < n) x[i] += tanhf(t[i]);
}

__global__ void k_gelu(float* __restrict__ h, int n) {
    int i = blockIdx.x * blockDim.x + threadIdx.x;
    if (i < n) {
        float v = h[i];
        h[i] = 0.5f * v * (1.0f + erff(v * 0.70710678118654752f));
    }
}

// per-(b,d) causal scan: s = cumsum(k*v); x += q * s * sigmoid(gate)
__global__ void k_scan_gate(const float* __restrict__ qkv,
                            const float* __restrict__ gate,
                            float* __restrict__ x) {
    int t = blockIdx.x * blockDim.x + threadIdx.x;   // 0..2047
    int b = t >> 9, d = t & 511;
    float s = 0.0f;
    for (int lp = 0; lp < LP; ++lp) {
        size_t r = (size_t)b * LP + lp;
        const float* q = qkv + r * (3 * DIM);
        float qq = q[d], kk = q[DIM + d], vv = q[2 * DIM + d];
        s += kk * vv;
        float gv = gate[r * DIM + d];
        float sg = 1.0f / (1.0f + expf(-gv));
        x[r * DIM + d] += qq * s * sg;
    }
}

// bc[o] = head_b[o] + sum_d dec_b[d]*head_w[d,o]
__global__ void k_head_bias(const float* __restrict__ dec_b,
                            const float* __restrict__ head_w,
                            const float* __restrict__ head_b,
                            float* __restrict__ bc) {
    int o = threadIdx.x;
    float s = head_b[o];
    for (int d = 0; d < DIM; ++d) s += dec_b[d] * head_w[(size_t)d * 256 + o];
    bc[o] = s;
}

// ---------------------------------------------------------------------------
extern "C" void kernel_launch(void* const* d_in, const int* in_sizes, int n_in,
                              void* d_out, int out_size, void* d_ws, size_t ws_size,
                              hipStream_t stream) {
    const int*   bytes_in = (const int*)  d_in[0];
    const float* byte_emb = (const float*)d_in[1];
    const float* enc_w    = (const float*)d_in[2];
    const float* enc_b    = (const float*)d_in[3];
    const float* hashp    = (const float*)d_in[4];
    const float* qkv_w    = (const float*)d_in[5];
    const float* qkv_b    = (const float*)d_in[6];
    const float* gate_w   = (const float*)d_in[7];
    const float* gate_b   = (const float*)d_in[8];
    const float* ln1_g    = (const float*)d_in[9];
    const float* ln1_b    = (const float*)d_in[10];
    const float* mlp_w1   = (const float*)d_in[11];
    const float* mlp_b1   = (const float*)d_in[12];
    const float* mlp_w2   = (const float*)d_in[13];
    const float* mlp_b2   = (const float*)d_in[14];
    const float* ln2_g    = (const float*)d_in[15];
    const float* ln2_b    = (const float*)d_in[16];
    const float* dec_w    = (const float*)d_in[17];
    const float* dec_b    = (const float*)d_in[18];
    const float* head_w   = (const float*)d_in[19];
    const float* head_b   = (const float*)d_in[20];
    float* logits = (float*)d_out;   // [4,8192,256] f32

    // ---- workspace carve-up ----
    char* wp = (char*)d_ws;
    auto alloc = [&](size_t bytes) -> char* {
        char* p = wp; wp += (bytes + 255) & ~(size_t)255; return p;
    };
    float* x     = (float*)alloc((size_t)ROWS * DIM * 4);        // 8 MB
    float* h     = (float*)alloc((size_t)ROWS * DIM * 4);        // 8 MB
    float* qkv   = (float*)alloc((size_t)ROWS * 3 * DIM * 4);    // 24 MB
    float* gbuf  = (float*)alloc((size_t)ROWS * DIM * 4);        // 8 MB
    float* mlph  = (float*)alloc((size_t)ROWS * 4 * DIM * 4);    // 32 MB
    float* Wc32  = (float*)alloc((size_t)DIM * 2048 * 4);        // 4 MB
    float* bc    = (float*)alloc(256 * 4);
    h16* A16     = (h16*)alloc((size_t)ROWS * 4096 * 2);         // 32 MB (max A)
    h16* encT    = (h16*)alloc((size_t)4096 * DIM * 2);
    h16* hashT   = (h16*)alloc((size_t)DIM * DIM * 2);
    h16* qkvT    = (h16*)alloc((size_t)NL * DIM * 3 * DIM * 2);
    h16* gateT   = (h16*)alloc((size_t)NL * DIM * DIM * 2);
    h16* w1T     = (h16*)alloc((size_t)NL * DIM * 4 * DIM * 2);
    h16* w2T     = (h16*)alloc((size_t)NL * 4 * DIM * DIM * 2);
    h16* headT   = (h16*)alloc((size_t)DIM * 256 * 2);
    h16* dec16   = (h16*)alloc((size_t)DIM * PP * DIM * 2);
    h16* WcT     = (h16*)alloc((size_t)2048 * DIM * 2);

    auto cvt = [&](const float* s, h16* d, int n) {
        k_cvt<<<(n + 255) / 256, 256, 0, stream>>>(s, d, n);
    };
    auto cvtT = [&](const float* s, h16* d, int K, int N) {
        k_cvt_t<<<(K * N + 255) / 256, 256, 0, stream>>>(s, d, K, N);
    };
    auto gemm = [&](const h16* A, int lda, const h16* Bt, int ldb,
                    float* C, int ldc, int M, int N, int K,
                    const float* bias, int bias_mod, int accf) {
        dim3 g(N / 128, M / 128);
        k_wmma_gemm<<<g, 256, 0, stream>>>(A, lda, Bt, ldb, C, ldc, K,
                                           bias, bias_mod, accf);
    };

    // ---- weight conversion (f32 -> f16, B matrices transposed) ----
    cvtT(enc_w, encT, 4096, DIM);
    cvtT(hashp, hashT, DIM, DIM);
    for (int i = 0; i < NL; ++i) {
        cvtT(qkv_w  + (size_t)i * DIM * 3 * DIM, qkvT  + (size_t)i * DIM * 3 * DIM, DIM, 3 * DIM);
        cvtT(gate_w + (size_t)i * DIM * DIM,     gateT + (size_t)i * DIM * DIM,     DIM, DIM);
        cvtT(mlp_w1 + (size_t)i * DIM * 4 * DIM, w1T   + (size_t)i * DIM * 4 * DIM, DIM, 4 * DIM);
        cvtT(mlp_w2 + (size_t)i * 4 * DIM * DIM, w2T   + (size_t)i * 4 * DIM * DIM, 4 * DIM, DIM);
    }
    cvtT(head_w, headT, DIM, 256);
    cvt(dec_w, dec16, DIM * PP * DIM);           // A-side, row-major [d, p*512+o2]

    // ---- fold decoder+head: Wc[d, p*256+o] = dec_w[d,p,:] @ head_w ----
    k_head_bias<<<1, 256, 0, stream>>>(dec_b, head_w, head_b, bc);
    for (int p = 0; p < PP; ++p)
        gemm(dec16 + p * DIM, PP * DIM, headT, DIM,
             Wc32 + p * 256, 2048, DIM, 256, DIM, nullptr, 1, 0);
    cvtT(Wc32, WcT, DIM, 2048);

    // ---- byte embed + patch encoder (GEMM over p*d) ----
    k_embed<<<(ROWS * 4096) / 256, 256, 0, stream>>>(bytes_in, byte_emb, A16);
    gemm(A16, 4096, encT, 4096, x, DIM, ROWS, DIM, 4096, enc_b, DIM, 0);

    // ---- hash projection: x += tanh(x @ hash_proj) ----
    cvt(x, A16, ROWS * DIM);
    gemm(A16, DIM, hashT, DIM, h, DIM, ROWS, DIM, DIM, nullptr, 1, 0);
    k_tanh_add<<<(ROWS * DIM) / 256, 256, 0, stream>>>(x, h, ROWS * DIM);

    // ---- transformer layers ----
    for (int i = 0; i < NL; ++i) {
        k_layernorm<<<ROWS, 256, 0, stream>>>(x, ln1_g + i * DIM, ln1_b + i * DIM, h);
        cvt(h, A16, ROWS * DIM);
        gemm(A16, DIM, qkvT + (size_t)i * DIM * 3 * DIM, DIM,
             qkv, 3 * DIM, ROWS, 3 * DIM, DIM, qkv_b + i * 3 * DIM, 3 * DIM, 0);
        gemm(A16, DIM, gateT + (size_t)i * DIM * DIM, DIM,
             gbuf, DIM, ROWS, DIM, DIM, gate_b + i * DIM, DIM, 0);
        k_scan_gate<<<8, 256, 0, stream>>>(qkv, gbuf, x);

        k_layernorm<<<ROWS, 256, 0, stream>>>(x, ln2_g + i * DIM, ln2_b + i * DIM, h);
        cvt(h, A16, ROWS * DIM);
        gemm(A16, DIM, w1T + (size_t)i * DIM * 4 * DIM, DIM,
             mlph, 4 * DIM, ROWS, 4 * DIM, DIM, mlp_b1 + i * 4 * DIM, 4 * DIM, 0);
        k_gelu<<<(ROWS * 4 * DIM) / 256, 256, 0, stream>>>(mlph, ROWS * 4 * DIM);
        cvt(mlph, A16, ROWS * 4 * DIM);
        gemm(A16, 4 * DIM, w2T + (size_t)i * 4 * DIM * DIM, 4 * DIM,
             x, DIM, ROWS, DIM, 4 * DIM, mlp_b2 + i * DIM, DIM, /*acc=*/1);
    }

    // ---- fused decoder+head: logits = x @ Wc + bc (layout == [B,L,256]) ----
    cvt(x, A16, ROWS * DIM);
    gemm(A16, DIM, WcT, DIM, logits, 2048, ROWS, 2048, DIM, bc, 256, 0);
}
